// SelfAttention_14370960572735
// MI455X (gfx1250) — matrix operand order repaired and checked by reading, not compiled, not run on hardware
//
#include <hip/hip_runtime.h>

// ---------------------------------------------------------------------------
// MI455X (gfx1250, wave32) fused self-attention, bf16 WMMA everywhere.
//   x(f32) -> bf16; W -> W^T bf16; QKV GEMMs (double-buffered K pipeline);
//   flash attention (online softmax, batched fragment loads); O GEMM -> f32.
// ---------------------------------------------------------------------------

typedef __bf16 bf16;
typedef __attribute__((ext_vector_type(16))) __bf16 v16bf;
typedef __attribute__((ext_vector_type(8)))  __bf16 v8bf;
typedef __attribute__((ext_vector_type(8)))  float  v8f;

union ABFrag { v16bf v; v8bf h[2]; };

#define B_SZ    2
#define T_SEQ   2048
#define DIM_C   4096
#define HQ      32
#define HKV     8
#define HD      128
#define M_TOT   (B_SZ * T_SEQ)          // 4096 rows
#define SCALE   0.08838834764831845f    // 1/sqrt(128)

__device__ __forceinline__ bf16 f2bf(float f) {
    unsigned u = __float_as_uint(f);
    u = u + 0x7fffu + ((u >> 16) & 1u);          // round-to-nearest-even
    unsigned short s = (unsigned short)(u >> 16);
    return __builtin_bit_cast(bf16, s);
}

__device__ __forceinline__ v8bf ld8(const bf16* p) { return *(const v8bf*)p; }

__device__ __forceinline__ v8f vzero() {
    v8f z;
#pragma unroll
    for (int i = 0; i < 8; ++i) z[i] = 0.0f;
    return z;
}

// -------------------------- conversion kernels -----------------------------

__global__ __launch_bounds__(256) void cvt_f32_bf16(const float* __restrict__ in,
                                                    bf16* __restrict__ out, size_t n) {
    size_t i = (size_t)blockIdx.x * blockDim.x + threadIdx.x;
    size_t stride = (size_t)gridDim.x * blockDim.x;
    for (; i < n; i += stride) out[i] = f2bf(in[i]);
}

// in: R x C f32 (row-major)  ->  out: C x R bf16 (row-major)   (R,C mult of 32)
__global__ __launch_bounds__(256) void transpose_f32_bf16(const float* __restrict__ in,
                                                          bf16* __restrict__ out,
                                                          int R, int C) {
    __shared__ float tile[32][33];
    int c0 = blockIdx.x * 32, r0 = blockIdx.y * 32;
    int tx = threadIdx.x & 31, ty = threadIdx.x >> 5;   // 32 x 8
#pragma unroll
    for (int i = 0; i < 32; i += 8)
        tile[ty + i][tx] = in[(size_t)(r0 + ty + i) * C + c0 + tx];
    __syncthreads();
#pragma unroll
    for (int i = 0; i < 32; i += 8)
        out[(size_t)(c0 + ty + i) * R + r0 + tx] = f2bf(tile[tx][ty + i]);
}

// Vb: (B*T, HKV*HD) bf16  ->  Vt: (B, HKV, HD, T) bf16
__global__ __launch_bounds__(128) void transpose_v(const bf16* __restrict__ Vb,
                                                   bf16* __restrict__ Vt) {
    int t = blockIdx.x;                  // 0..T-1
    int bh = blockIdx.y;                 // 0..B*HKV-1
    int b = bh >> 3, hk = bh & 7;
    int d = threadIdx.x;                 // 0..127  (coalesced read)
    bf16 v = Vb[(size_t)(b * T_SEQ + t) * (HKV * HD) + hk * HD + d];
    Vt[((size_t)bh * HD + d) * T_SEQ + t] = v;
}

// ------------------------------- GEMM --------------------------------------
// C[M,N] = A[M,K](bf16 rm) * Bt[N,K](bf16 rm)^T + bias[N]
// block = 256 thr = 8 waves (2x4), block tile 128x128, wave tile 64x32.
// K pipelined in steps of 64 with ping-pong fragment buffers.
template <bool OUT_F32>
__global__ __launch_bounds__(256) void gemm_bf16(const bf16* __restrict__ A,
                                                 const bf16* __restrict__ Bt,
                                                 const float* __restrict__ bias,
                                                 void* __restrict__ Cout,
                                                 int M, int N, int K) {
    const int lane = threadIdx.x & 31;
    const int wave = threadIdx.x >> 5;
    const int m0 = blockIdx.y * 128 + (wave >> 2) * 64;
    const int n0 = blockIdx.x * 128 + (wave & 3) * 32;
    const int mrow = lane & 15;
    const int aklo = (lane >> 4) * 8;    // A frag: K {klo..klo+7} U {16+klo..}
    const int bklo = (lane >> 4) * 8;    // B frag: K {klo..klo+7} U {16+klo..}
    const int rbase = (lane >> 4) * 8;   // C frag row base

    // loop-invariant per-fragment row pointers
    const bf16* aP[4];
    const bf16* bP[2];
#pragma unroll
    for (int mt = 0; mt < 4; ++mt)
        aP[mt] = A + (size_t)(m0 + mt * 16 + mrow) * K + aklo;
#pragma unroll
    for (int nt = 0; nt < 2; ++nt)
        bP[nt] = Bt + (size_t)(n0 + nt * 16 + mrow) * K + bklo;

    v8f acc[4][2];
#pragma unroll
    for (int mt = 0; mt < 4; ++mt)
#pragma unroll
        for (int nt = 0; nt < 2; ++nt) acc[mt][nt] = vzero();

    ABFrag a0[4], b0[2], a1[4], b1[2];

#define LOAD_STAGE(AA, BB, KK)                                            \
    do {                                                                  \
        _Pragma("unroll") for (int mt = 0; mt < 4; ++mt) {                \
            (AA)[mt].h[0] = ld8(aP[mt] + (KK));                           \
            (AA)[mt].h[1] = ld8(aP[mt] + (KK) + 16);                      \
        }                                                                 \
        _Pragma("unroll") for (int nt = 0; nt < 2; ++nt) {                \
            (BB)[nt].h[0] = ld8(bP[nt] + (KK));                           \
            (BB)[nt].h[1] = ld8(bP[nt] + (KK) + 16);                      \
        }                                                                 \
    } while (0)

#define MMA_STAGE(AA, BB)                                                 \
    do {                                                                  \
        _Pragma("unroll") for (int mt = 0; mt < 4; ++mt)                  \
            _Pragma("unroll") for (int nt = 0; nt < 2; ++nt)              \
                acc[mt][nt] = __builtin_amdgcn_wmma_f32_16x16x32_bf16(    \
                    false, (AA)[mt].v, false, (BB)[nt].v, (short)0,       \
                    acc[mt][nt], false, false);                           \
    } while (0)

    LOAD_STAGE(a0, b0, 0);
    for (int k0 = 0; k0 < K; k0 += 64) {
        LOAD_STAGE(a1, b1, k0 + 32);        // stage n+1 loads in flight
        MMA_STAGE(a0, b0);                  // stage n compute
        if (k0 + 64 < K) LOAD_STAGE(a0, b0, k0 + 64);
        MMA_STAGE(a1, b1);
    }
#undef LOAD_STAGE
#undef MMA_STAGE

#pragma unroll
    for (int nt = 0; nt < 2; ++nt) {
        const int col = n0 + nt * 16 + mrow;
        const float bv = bias[col];
#pragma unroll
        for (int mt = 0; mt < 4; ++mt) {
#pragma unroll
            for (int j = 0; j < 8; ++j) {
                const int row = m0 + mt * 16 + rbase + j;
                const float v = acc[mt][nt][j] + bv;
                if constexpr (OUT_F32)
                    ((float*)Cout)[(size_t)row * N + col] = v;
                else
                    ((bf16*)Cout)[(size_t)row * N + col] = f2bf(v);
            }
        }
    }
}

// --------------------------- flash attention -------------------------------
// grid (T/128, HQ, B), block 256 = 8 waves; each wave owns 16 query rows.
__global__ __launch_bounds__(256) void flash_attn(const bf16* __restrict__ Qb,
                                                  const bf16* __restrict__ Kb,
                                                  const bf16* __restrict__ Vt,
                                                  bf16* __restrict__ Ctx) {
    const int lane = threadIdx.x & 31;
    const int wave = threadIdx.x >> 5;
    const int h = blockIdx.y, b = blockIdx.z;
    const int hk = h >> 2;                          // GQA: 4 Q heads / KV head
    const int q0 = blockIdx.x * 128 + wave * 16;
    const int mrow = lane & 15;
    const int aklo = (lane >> 4) * 8;
    const int bklo = (lane >> 4) * 8;
    const int rbase = (lane >> 4) * 8;

    __shared__ float S_s[8][16][33];
    __shared__ __align__(16) bf16 P_s[8][16][32];
    __shared__ float al_s[8][16];
    __shared__ float li_s[8][16];

    // resident Q fragments (16 rows x 128 d) = 4 A-frags
    ABFrag qf[4];
#pragma unroll
    for (int dt = 0; dt < 4; ++dt) {
        const bf16* p = Qb + (size_t)(b * T_SEQ + q0 + mrow) * (HQ * HD) +
                        h * HD + dt * 32 + aklo;
        qf[dt].h[0] = ld8(p);
        qf[dt].h[1] = ld8(p + 16);
    }

    // loop-invariant base pointers (offset by kk0 each iteration)
    const bf16* kBase[2];
#pragma unroll
    for (int nt = 0; nt < 2; ++nt)
        kBase[nt] = Kb + (size_t)(b * T_SEQ + nt * 16 + mrow) * (HKV * HD) +
                    hk * HD + bklo;
    const bf16* vBase[8];
#pragma unroll
    for (int nt2 = 0; nt2 < 8; ++nt2)
        vBase[nt2] = Vt +
            ((size_t)((b * HKV + hk) * HD + nt2 * 16 + mrow)) * T_SEQ + bklo;

    v8f o[8];
#pragma unroll
    for (int i = 0; i < 8; ++i) o[i] = vzero();
    float m_r = -3.0e38f, l_r = 0.0f;               // valid on lanes 0..15

    for (int kk0 = 0; kk0 < T_SEQ; kk0 += 32) {
        // ---- S = Q * K^T (16x32), 8 WMMAs; K frags batched per n-tile ----
        v8f s[2];
        s[0] = vzero();
        s[1] = vzero();
#pragma unroll
        for (int nt = 0; nt < 2; ++nt) {
            ABFrag kf[4];
#pragma unroll
            for (int dt = 0; dt < 4; ++dt) {
                const bf16* p = kBase[nt] + (size_t)kk0 * (HKV * HD) + dt * 32;
                kf[dt].h[0] = ld8(p);
                kf[dt].h[1] = ld8(p + 16);
            }
#pragma unroll
            for (int dt = 0; dt < 4; ++dt)
                s[nt] = __builtin_amdgcn_wmma_f32_16x16x32_bf16(
                    false, qf[dt].v, false, kf[dt].v, (short)0, s[nt],
                    false, false);
        }

        // ---- issue all V-fragment loads now: they overlap softmax+barriers
        ABFrag vf[8];
#pragma unroll
        for (int nt2 = 0; nt2 < 8; ++nt2) {
            const bf16* vp = vBase[nt2] + kk0;
            vf[nt2].h[0] = ld8(vp);
            vf[nt2].h[1] = ld8(vp + 16);
        }

        // dump S tile to per-wave LDS region
#pragma unroll
        for (int nt = 0; nt < 2; ++nt)
#pragma unroll
            for (int j = 0; j < 8; ++j)
                S_s[wave][rbase + j][nt * 16 + mrow] = s[nt][j];
        __syncthreads();

        // ---- online softmax: lane r (<16) handles row r ----
        if (lane < 16) {
            float tm = -3.0e38f;
#pragma unroll
            for (int c = 0; c < 32; ++c)
                tm = fmaxf(tm, S_s[wave][lane][c] * SCALE);
            const float mn = fmaxf(m_r, tm);
            const float al = __expf(m_r - mn);
            float ls = 0.0f;
#pragma unroll
            for (int c = 0; c < 32; ++c) {
                const float pv = __expf(S_s[wave][lane][c] * SCALE - mn);
                ls += pv;
                P_s[wave][lane][c] = f2bf(pv);
            }
            l_r = l_r * al + ls;
            m_r = mn;
            al_s[wave][lane] = al;
        }
        __syncthreads();

        // ---- rescale running output by alpha ----
#pragma unroll
        for (int j = 0; j < 8; ++j) {
            const float al = al_s[wave][rbase + j];
#pragma unroll
            for (int nt2 = 0; nt2 < 8; ++nt2) o[nt2][j] *= al;
        }

        // ---- O += P * V  (16x128), 8 WMMAs with preloaded V frags ----
        ABFrag pf;
        {
            const bf16* pp = &P_s[wave][mrow][aklo];
            pf.h[0] = ld8(pp);
            pf.h[1] = ld8(pp + 16);
        }
#pragma unroll
        for (int nt2 = 0; nt2 < 8; ++nt2)
            o[nt2] = __builtin_amdgcn_wmma_f32_16x16x32_bf16(
                false, pf.v, false, vf[nt2].v, (short)0, o[nt2], false, false);
        __syncthreads();   // protect S_s / P_s for next tile
    }

    if (lane < 16) li_s[wave][lane] = 1.0f / l_r;
    __syncthreads();

#pragma unroll
    for (int nt2 = 0; nt2 < 8; ++nt2) {
#pragma unroll
        for (int j = 0; j < 8; ++j) {
            const int r = rbase + j;
            const float val = o[nt2][j] * li_s[wave][r];
            Ctx[(size_t)(b * T_SEQ + q0 + r) * (HQ * HD) + h * HD + nt2 * 16 + mrow] =
                f2bf(val);
        }
    }
}

// ------------------------------ launcher -----------------------------------

extern "C" void kernel_launch(void* const* d_in, const int* in_sizes, int n_in,
                              void* d_out, int out_size, void* d_ws, size_t ws_size,
                              hipStream_t stream) {
    (void)in_sizes; (void)n_in; (void)out_size; (void)ws_size;
    const float* x   = (const float*)d_in[0];
    /* d_in[1] = mask: reference drops it (non-in-place masked_fill) -> ignored */
    const float* w_q = (const float*)d_in[2];
    const float* b_q = (const float*)d_in[3];
    const float* w_k = (const float*)d_in[4];
    const float* b_k = (const float*)d_in[5];
    const float* w_v = (const float*)d_in[6];
    const float* b_v = (const float*)d_in[7];
    const float* w_o = (const float*)d_in[8];
    const float* b_o = (const float*)d_in[9];
    float* out = (float*)d_out;

    char* ws = (char*)d_ws;
    constexpr size_t MB = 1024ull * 1024ull;
    bf16* xb  = (bf16*)(ws +   0 * MB);   // 4096x4096           32 MB
    bf16* wqT = (bf16*)(ws +  32 * MB);   // 4096x4096 (N,K)     32 MB
    bf16* wkT = (bf16*)(ws +  64 * MB);   // 1024x4096            8 MB
    bf16* wvT = (bf16*)(ws +  72 * MB);   // 1024x4096            8 MB
    bf16* woT = (bf16*)(ws +  80 * MB);   // 4096x4096           32 MB
    bf16* Qb  = (bf16*)(ws + 112 * MB);   // 4096x4096           32 MB
    bf16* Kb  = (bf16*)(ws + 144 * MB);   // 4096x1024            8 MB
    bf16* Vb  = (bf16*)(ws + 152 * MB);   // 4096x1024            8 MB
    bf16* Vt  = (bf16*)(ws + 160 * MB);   // (B,HKV,HD,T)         8 MB
    bf16* Ctx = (bf16*)(ws + 168 * MB);   // 4096x4096           32 MB
                                          // total 200 MB

    // 1) precision conversion / weight transposes (one streaming pass)
    cvt_f32_bf16<<<8192, 256, 0, stream>>>(x, xb, (size_t)M_TOT * DIM_C);
    transpose_f32_bf16<<<dim3(DIM_C / 32, DIM_C / 32), 256, 0, stream>>>(w_q, wqT, DIM_C, DIM_C);
    transpose_f32_bf16<<<dim3((HKV * HD) / 32, DIM_C / 32), 256, 0, stream>>>(w_k, wkT, DIM_C, HKV * HD);
    transpose_f32_bf16<<<dim3((HKV * HD) / 32, DIM_C / 32), 256, 0, stream>>>(w_v, wvT, DIM_C, HKV * HD);
    transpose_f32_bf16<<<dim3(DIM_C / 32, DIM_C / 32), 256, 0, stream>>>(w_o, woT, DIM_C, DIM_C);

    // 2) QKV projections (WMMA bf16, f32 accum, bf16 out)
    gemm_bf16<false><<<dim3(DIM_C / 128, M_TOT / 128), 256, 0, stream>>>(
        xb, wqT, b_q, Qb, M_TOT, DIM_C, DIM_C);
    gemm_bf16<false><<<dim3((HKV * HD) / 128, M_TOT / 128), 256, 0, stream>>>(
        xb, wkT, b_k, Kb, M_TOT, HKV * HD, DIM_C);
    gemm_bf16<false><<<dim3((HKV * HD) / 128, M_TOT / 128), 256, 0, stream>>>(
        xb, wvT, b_v, Vb, M_TOT, HKV * HD, DIM_C);

    // 3) V -> V^T per (batch, kv-head) so PV B-fragments are contiguous
    transpose_v<<<dim3(T_SEQ, B_SZ * HKV), 128, 0, stream>>>(Vb, Vt);

    // 4) flash attention (unmasked, per reference)
    flash_attn<<<dim3(T_SEQ / 128, HQ, B_SZ), 256, 0, stream>>>(Qb, Kb, Vt, Ctx);

    // 5) output projection, f32 out + bias
    gemm_bf16<true><<<dim3(DIM_C / 128, M_TOT / 128), 256, 0, stream>>>(
        Ctx, woT, b_o, out, M_TOT, DIM_C, DIM_C);
}
